// GroupedQueryAttention_70016556860001
// MI455X (gfx1250) — compile-verified, matrix operand order checked by reading, not run on hardware
//
#include <hip/hip_runtime.h>
#include <hip/hip_bf16.h>

typedef __attribute__((ext_vector_type(16))) __bf16 v16bf;
typedef __attribute__((ext_vector_type(8)))  __bf16 v8bf;
typedef __attribute__((ext_vector_type(8)))  float  v8f;
typedef __attribute__((ext_vector_type(4)))  unsigned int v4u;
typedef __attribute__((ext_vector_type(4)))  int v4i;
typedef __attribute__((ext_vector_type(8)))  int v8i;

static constexpr int EMB = 2048, NH = 16, NG = 4, HD = 128;
static constexpr int BB = 2, TT = 2048;
static constexpr int NTOT = 3072;   // Q(2048) | K(512) | V(512)

__device__ __forceinline__ v8f wmma_bf16(v16bf a, v16bf b, v8f c) {
  return __builtin_amdgcn_wmma_f32_16x16x32_bf16(false, a, false, b, (short)0, c,
                                                 false, false);
}

// load 16 fp32 and convert to a bf16 A/B fragment half-row
__device__ __forceinline__ v16bf cvt16(const float* __restrict__ p) {
  v16bf r;
#pragma unroll
  for (int i = 0; i < 16; ++i) r[i] = (__bf16)p[i];
  return r;
}

// 32-byte bf16 fragment load as two 16B chunks (16B-aligned LDS rows OK)
__device__ __forceinline__ v16bf ld16bf(const __bf16* p) {
  union { v16bf v; v8bf h[2]; } u;
  u.h[0] = *(const v8bf*)(p);
  u.h[1] = *(const v8bf*)(p + 8);
  return u.v;
}

// ---------------------------------------------------------------------------
// TDM: DMA a 64x128 bf16 K-tile (row stride 128 elems in global) into LDS,
// with D# padding of 8 dwords (16 bf16) after every 64 dwords (one row) so
// the LDS image matches kTile[64][HD+16] and B-fragment reads are
// bank-conflict free.  Tracked with TENSORcnt.
// ---------------------------------------------------------------------------
__device__ __forceinline__ void tdm_load_k(const __bf16* gsrc,
                                           unsigned int lds_off) {
  const unsigned long long ga = (unsigned long long)(size_t)gsrc;
  // group0: count=1 | lds_addr | global_addr[56:0] | type=2 ("image")
  v4u g0 = { 1u, lds_off, (unsigned int)ga,
             (unsigned int)((ga >> 32) & 0x01FFFFFFu) | (2u << 30) };
  // group1: data_size=2B, pad_enable, pad_interval=64dw, pad_amount=8dw,
  //         tensor_dim0=128, tensor_dim1=2048, tile_dim0=128, tile_dim1=64,
  //         tensor_dim0_stride=128
  v8i g1 = { (int)((1u << 16) | (1u << 20) | (5u << 22) | (7u << 25)),
             (int)(128u << 16),   // tensor_dim0 low half  (bits 79:48)
             (int)(2048u << 16),  // tensor_dim1 low half  (bits 111:80)
             (int)(128u << 16),   // tile_dim0             (bits 127:112)
             64,                  // tile_dim1             (bits 143:128)
             128,                 // tensor_dim0_stride    (bits 207:160)
             0, 0 };
  v4i g2 = { 0, 0, 0, 0 };        // 2D: groups 2/3 unused (all-zero = inert)
  v4i g3 = { 0, 0, 0, 0 };
#if defined(__clang_major__) && __clang_major__ >= 23
  v8i g4 = { 0, 0, 0, 0, 0, 0, 0, 0 };
  __builtin_amdgcn_tensor_load_to_lds(g0, g1, g2, g3, g4, 0);
#else
  __builtin_amdgcn_tensor_load_to_lds(g0, g1, g2, g3, 0);
#endif
}

// ---------------------------------------------------------------------------
// Kernel 1: fused QKV projection + RoPE.  C = x @ [Wq|Wk|Wv]^T, bf16 out.
// Block = 256 threads (8 waves). Block tile 128(M) x 128(N); wave tile 32x64.
// ---------------------------------------------------------------------------
__global__ __launch_bounds__(256) void qkv_rope_kernel(
    const float* __restrict__ x, const float* __restrict__ Wq,
    const float* __restrict__ Wk, const float* __restrict__ Wv,
    __bf16* __restrict__ qws, __bf16* __restrict__ kws,
    __bf16* __restrict__ vws) {
  const int lane = threadIdx.x & 31;
  const int wave = threadIdx.x >> 5;
  const int wm = wave & 3, wn = wave >> 2;
  const int laneN = lane & 15, grp = lane >> 4, laneK = grp << 4;
  const int mBase = blockIdx.y * 128 + wm * 32;
  const int nBase = blockIdx.x * 128 + wn * 64;

  const float* Wsel;
  int nOff, mode;                        // 0=Q, 1=K, 2=V  (uniform per block)
  if (blockIdx.x < 16)      { Wsel = Wq; nOff = 0;    mode = 0; }
  else if (blockIdx.x < 20) { Wsel = Wk; nOff = 2048; mode = 1; }
  else                      { Wsel = Wv; nOff = 2560; mode = 2; }

  v8f acc[2][4];
#pragma unroll
  for (int s = 0; s < 2; ++s)
#pragma unroll
    for (int j = 0; j < 4; ++j)
#pragma unroll
      for (int i = 0; i < 8; ++i) acc[s][j][i] = 0.f;

  for (int k0 = 0; k0 < EMB; k0 += 32) {
    v16bf af[2], bf[4];
#pragma unroll
    for (int s = 0; s < 2; ++s)
      af[s] = cvt16(x + (size_t)(mBase + s * 16 + laneN) * EMB + k0 + laneK);
#pragma unroll
    for (int j = 0; j < 4; ++j)
      bf[j] = cvt16(Wsel + (size_t)(nBase + j * 16 + laneN - nOff) * EMB + k0 + laneK);
#pragma unroll
    for (int s = 0; s < 2; ++s)
#pragma unroll
      for (int j = 0; j < 4; ++j)
        acc[s][j] = wmma_bf16(af[s], bf[j], acc[s][j]);
  }

  const float NEG_LN1E4_DIV64 = -0.14391156f;  // -ln(10000)/64

#pragma unroll
  for (int s = 0; s < 2; ++s) {
#pragma unroll
    for (int j = 0; j < 4; ++j) {
      const int n = nBase + j * 16 + laneN;
#pragma unroll
      for (int i = 0; i < 8; ++i) {
        const int m = mBase + s * 16 + 8 * grp + i;  // C layout: row = 8*grp+i
        const int b = m >> 11, t = m & (TT - 1);
        float v = acc[s][j][i];
        if (mode < 2) {  // RoPE on Q and K: pair partner is the adjacent lane
          const float partner = __shfl_xor(v, 1, 32);
          const int hd = n & (HD - 1);
          const float ang = (float)t * __expf((float)(hd >> 1) * NEG_LN1E4_DIV64);
          float sn, cs;
          __sincosf(ang, &sn, &cs);
          v = (hd & 1) ? (partner * sn + v * cs) : (v * cs - partner * sn);
        }
        if (mode == 0) {
          const int h = n >> 7, hd = n & 127;
          qws[(((size_t)b * NH + h) * TT + t) * HD + hd] = (__bf16)v;
        } else if (mode == 1) {
          const int nn = n - 2048, g = nn >> 7, hd = nn & 127;
          kws[(((size_t)b * NG + g) * TT + t) * HD + hd] = (__bf16)v;
        } else {
          const int nn = n - 2560, g = nn >> 7, hd = nn & 127;
          vws[(((size_t)b * NG + g) * TT + t) * HD + hd] = (__bf16)v;
        }
      }
    }
  }
}

// ---------------------------------------------------------------------------
// Kernel 2: flash attention with TDM-double-buffered K tiles and prefetched
// V^T tiles. Block = (b, head, 128 query rows), 8 waves; each wave owns 16
// query rows.
// ---------------------------------------------------------------------------
__global__ __launch_bounds__(256) void attn_kernel(
    const __bf16* __restrict__ qws, const __bf16* __restrict__ kws,
    const __bf16* __restrict__ vws, __bf16* __restrict__ aws) {
  __shared__ __attribute__((aligned(32))) __bf16 kTile[2][64][HD + 16]; // 36 KB
  __shared__ __attribute__((aligned(32))) __bf16 vT[2][HD][64 + 16];   // 40 KB
  __shared__ __attribute__((aligned(16))) float  pScr[8][16][64 + 4];  // 34 KB

  const int lane = threadIdx.x & 31, wave = threadIdx.x >> 5;
  const int laneN = lane & 15, grp = lane >> 4, laneK = grp << 4;
  const int b = blockIdx.z, h = blockIdx.y, g = h >> 2;
  const int qbase = blockIdx.x * 128 + wave * 16;

  const __bf16* Qh = qws + ((size_t)(b * NH + h) * TT) * HD;
  const __bf16* Kh = kws + ((size_t)(b * NG + g) * TT) * HD;
  const __bf16* Vh = vws + ((size_t)(b * NG + g) * TT) * HD;

  // Q A-fragments for this wave's 16 rows (K = 128 in 4 chunks of 32)
  v16bf qf[4];
#pragma unroll
  for (int kk = 0; kk < 4; ++kk)
    qf[kk] = ld16bf(Qh + (size_t)(qbase + laneN) * HD + kk * 32 + laneK);

  v8f o[8];
  float rm[8], rl[8];
#pragma unroll
  for (int ot = 0; ot < 8; ++ot)
#pragma unroll
    for (int i = 0; i < 8; ++i) o[ot][i] = 0.f;
#pragma unroll
  for (int i = 0; i < 8; ++i) { rm[i] = -3.0e38f; rl[i] = 0.f; }

  const float scale = 0.08838834764f;  // 1/sqrt(128)
  const int nkt = (blockIdx.x + 1) * 2;     // causal: only tiles <= block qmax
  const int cr = threadIdx.x >> 2;          // V staging row 0..63
  const int cc = (threadIdx.x & 3) * 32;    // V staging col segment

  // prologue: issue TDM for K tile 0 (wave 0) + stage V^T tile 0 (all waves)
  if (wave == 0) tdm_load_k(Kh, (unsigned int)(size_t)&kTile[0][0][0]);
  {
    const __bf16* vs = Vh + (size_t)cr * HD + cc;
#pragma unroll
    for (int u = 0; u < 32; ++u) vT[0][cc + u][cr] = vs[u];
  }

  for (int kt = 0; kt < nkt; ++kt) {
    const int buf = kt & 1;
    const int kbase = kt * 64;

    // prefetch next tile into the other buffer while this one is consumed
    if (kt + 1 < nkt) {
      if (wave == 0)
        tdm_load_k(Kh + (size_t)(kbase + 64) * HD,
                   (unsigned int)(size_t)&kTile[buf ^ 1][0][0]);
      const __bf16* vs = Vh + (size_t)(kbase + 64 + cr) * HD + cc;
#pragma unroll
      for (int u = 0; u < 32; ++u) vT[buf ^ 1][cc + u][cr] = vs[u];
    }
    if (wave == 0) {  // current K tile complete (1 prefetch may stay in flight)
      if (kt + 1 < nkt) __builtin_amdgcn_s_wait_tensorcnt(1);
      else              __builtin_amdgcn_s_wait_tensorcnt(0);
    }
    __syncthreads();

    // S = Q @ K^T : 4 tiles of 16x16, K-dim 128
    v8f s[4];
#pragma unroll
    for (int j = 0; j < 4; ++j) {
#pragma unroll
      for (int i = 0; i < 8; ++i) s[j][i] = 0.f;
#pragma unroll
      for (int kk = 0; kk < 4; ++kk)
        s[j] = wmma_bf16(qf[kk],
                         ld16bf(&kTile[buf][j * 16 + laneN][kk * 32 + laneK]),
                         s[j]);
    }
    // causal mask + scale (lane knows n = key col, vgpr index knows m)
#pragma unroll
    for (int j = 0; j < 4; ++j) {
      const int key = kbase + j * 16 + laneN;
#pragma unroll
      for (int i = 0; i < 8; ++i) {
        const int qr = qbase + 8 * grp + i;
        s[j][i] = (key <= qr) ? s[j][i] * scale : -3.0e38f;
      }
    }
    // online softmax: row reductions across the 16 lanes of each half-wave
    float tmax[8];
#pragma unroll
    for (int i = 0; i < 8; ++i)
      tmax[i] = fmaxf(fmaxf(s[0][i], s[1][i]), fmaxf(s[2][i], s[3][i]));
#pragma unroll
    for (int xm = 1; xm < 16; xm <<= 1)
#pragma unroll
      for (int i = 0; i < 8; ++i)
        tmax[i] = fmaxf(tmax[i], __shfl_xor(tmax[i], xm, 32));
    float alpha[8], rs[8];
#pragma unroll
    for (int i = 0; i < 8; ++i) {
      const float nm = fmaxf(rm[i], tmax[i]);
      alpha[i] = __expf(rm[i] - nm);
      rm[i] = nm;
      rs[i] = 0.f;
    }
#pragma unroll
    for (int j = 0; j < 4; ++j)
#pragma unroll
      for (int i = 0; i < 8; ++i) {
        const float p = __expf(s[j][i] - rm[i]);
        s[j][i] = p;
        rs[i] += p;
      }
#pragma unroll
    for (int xm = 1; xm < 16; xm <<= 1)
#pragma unroll
      for (int i = 0; i < 8; ++i) rs[i] += __shfl_xor(rs[i], xm, 32);
#pragma unroll
    for (int i = 0; i < 8; ++i) rl[i] = rl[i] * alpha[i] + rs[i];
#pragma unroll
    for (int ot = 0; ot < 8; ++ot)
#pragma unroll
      for (int i = 0; i < 8; ++i) o[ot][i] *= alpha[i];

    // reshape P: C-fragment layout -> A-fragment layout via per-wave scratch
#pragma unroll
    for (int j = 0; j < 4; ++j)
#pragma unroll
      for (int i = 0; i < 8; ++i)
        pScr[wave][8 * grp + i][j * 16 + laneN] = s[j][i];
    asm volatile("s_wait_dscnt 0" ::: "memory");  // same-wave LDS RAW
    v16bf pf[2];
#pragma unroll
    for (int f = 0; f < 2; ++f) {
      const float* pr = &pScr[wave][laneN][f * 32 + laneK];
      v16bf tv;
#pragma unroll
      for (int u = 0; u < 16; ++u) tv[u] = (__bf16)pr[u];
      pf[f] = tv;
    }
    // O += P @ V  (B-frag columns are head dims; vT gives contiguous k reads)
#pragma unroll
    for (int ot = 0; ot < 8; ++ot)
#pragma unroll
      for (int f = 0; f < 2; ++f)
        o[ot] = wmma_bf16(pf[f],
                          ld16bf(&vT[buf][ot * 16 + laneN][f * 32 + laneK]),
                          o[ot]);
    __syncthreads();
  }

  // epilogue: O /= l, store bf16 as (b, t, h*HD + d)
  float invl[8];
#pragma unroll
  for (int i = 0; i < 8; ++i) invl[i] = 1.0f / rl[i];
#pragma unroll
  for (int ot = 0; ot < 8; ++ot)
#pragma unroll
    for (int i = 0; i < 8; ++i) {
      const int t = qbase + 8 * grp + i;
      const int d = ot * 16 + laneN;
      aws[((size_t)b * TT + t) * EMB + h * HD + d] = (__bf16)(o[ot][i] * invl[i]);
    }
}

// ---------------------------------------------------------------------------
// Kernel 3: output projection.  out = attn(bf16) @ Wo^T, fp32 stores.
// ---------------------------------------------------------------------------
__global__ __launch_bounds__(256) void out_proj_kernel(
    const __bf16* __restrict__ aws, const float* __restrict__ Wo,
    float* __restrict__ out) {
  const int lane = threadIdx.x & 31, wave = threadIdx.x >> 5;
  const int wm = wave & 3, wn = wave >> 2;
  const int laneN = lane & 15, grp = lane >> 4, laneK = grp << 4;
  const int mBase = blockIdx.y * 128 + wm * 32;
  const int nBase = blockIdx.x * 128 + wn * 64;

  v8f acc[2][4];
#pragma unroll
  for (int s = 0; s < 2; ++s)
#pragma unroll
    for (int j = 0; j < 4; ++j)
#pragma unroll
      for (int i = 0; i < 8; ++i) acc[s][j][i] = 0.f;

  for (int k0 = 0; k0 < EMB; k0 += 32) {
    v16bf af[2], bf[4];
#pragma unroll
    for (int s = 0; s < 2; ++s)
      af[s] = ld16bf(aws + (size_t)(mBase + s * 16 + laneN) * EMB + k0 + laneK);
#pragma unroll
    for (int j = 0; j < 4; ++j)
      bf[j] = cvt16(Wo + (size_t)(nBase + j * 16 + laneN) * EMB + k0 + laneK);
#pragma unroll
    for (int s = 0; s < 2; ++s)
#pragma unroll
      for (int j = 0; j < 4; ++j)
        acc[s][j] = wmma_bf16(af[s], bf[j], acc[s][j]);
  }
#pragma unroll
  for (int s = 0; s < 2; ++s)
#pragma unroll
    for (int j = 0; j < 4; ++j)
#pragma unroll
      for (int i = 0; i < 8; ++i)
        out[(size_t)(mBase + s * 16 + 8 * grp + i) * EMB + nBase + j * 16 + laneN] =
            acc[s][j][i];
}

// ---------------------------------------------------------------------------
extern "C" void kernel_launch(void* const* d_in, const int* in_sizes, int n_in,
                              void* d_out, int out_size, void* d_ws,
                              size_t ws_size, hipStream_t stream) {
  (void)in_sizes; (void)n_in; (void)out_size; (void)ws_size;
  const float* x  = (const float*)d_in[0];
  const float* Wq = (const float*)d_in[1];
  const float* Wk = (const float*)d_in[2];
  const float* Wv = (const float*)d_in[3];
  const float* Wo = (const float*)d_in[4];

  __bf16* qws = (__bf16*)d_ws;                        // 16 MB
  __bf16* kws = qws + (size_t)BB * NH * TT * HD;      //  4 MB
  __bf16* vws = kws + (size_t)BB * NG * TT * HD;      //  4 MB
  __bf16* aws = vws + (size_t)BB * NG * TT * HD;      // 16 MB

  dim3 blk(256);
  qkv_rope_kernel<<<dim3(NTOT / 128, (BB * TT) / 128), blk, 0, stream>>>(
      x, Wq, Wk, Wv, qws, kws, vws);
  attn_kernel<<<dim3(TT / 128, NH, BB), blk, 0, stream>>>(qws, kws, vws, aws);
  out_proj_kernel<<<dim3(EMB / 128, (BB * TT) / 128), blk, 0, stream>>>(
      aws, Wo, (float*)d_out);
}
